// JaxLSTM_22256520528303
// MI455X (gfx1250) — matrix operand order repaired and checked by reading, not compiled, not run on hardware
//
#include <hip/hip_runtime.h>

// ---------- types ----------
typedef __bf16  v16bf __attribute__((ext_vector_type(16)));
typedef __bf16  v8bf  __attribute__((ext_vector_type(8)));
typedef float   v8f   __attribute__((ext_vector_type(8)));
typedef unsigned int u32x4 __attribute__((ext_vector_type(4)));
typedef int          i32x8 __attribute__((ext_vector_type(8)));
typedef int          i32x4 __attribute__((ext_vector_type(4)));

union Frag { v16bf v; v8bf h[2]; };

// ---------- constants ----------
#define BS    256
#define HL_F  8192      // 256*32 flattened history
#define H     1024
#define G4    4096      // 4*H
#define T     64
#define AD    16
#define KEXT  1056      // 1024 + 16 (action) + 16 (zero pad)
#define OUTD  16
#define NBLK  128       // persistent grid: 4 row-blocks x 32 col-blocks
#define LDSB_BYTES (128 * KEXT * 2)   // 270,336 B  (<= 320KB WGP LDS)

// ---------- scalar helpers ----------
__device__ __forceinline__ __bf16 f2bf(float f) {
    unsigned u = __builtin_bit_cast(unsigned, f);
    unsigned r = u + 0x7FFFu + ((u >> 16) & 1u);
    unsigned short h = (unsigned short)(r >> 16);
    return __builtin_bit_cast(__bf16, h);
}
__device__ __forceinline__ float bf2f(__bf16 b) {
    unsigned short h = __builtin_bit_cast(unsigned short, b);
    unsigned u = ((unsigned)h) << 16;
    return __builtin_bit_cast(float, u);
}
__device__ __forceinline__ float sigmoid_f(float x) {
    return 1.0f / (1.0f + __expf(-x));
}
__device__ __forceinline__ float tanh_f(float x) {
    x = fminf(15.0f, fmaxf(-15.0f, x));
    float e = __expf(2.0f * x);
    return (e - 1.0f) / (e + 1.0f);
}

// ---------- fragment loaders (bf16, 16x16x32 WMMA) ----------
// A 16x32 (MxK), row-major src: lane l: M=l&15; K chunks [8*sel..+7] and [16+8*sel..+7]
__device__ __forceinline__ v16bf load_fragA(const __bf16* __restrict__ base, int ld, int lane) {
    int m = lane & 15, sel = (lane >> 4) & 1;
    const __bf16* p = base + (long long)m * ld + sel * 8;
    Frag f;
    f.h[0] = *(const v8bf*)(p);
    f.h[1] = *(const v8bf*)(p + 16);
    return f.v;
}
// B 32x16 (KxN) from N-major (transposed) src: lane l: N=l&15; K chunks [8*sel..+7] and [16+8*sel..+7]
__device__ __forceinline__ v16bf load_fragB(const __bf16* __restrict__ base, int ld, int lane) {
    int n = lane & 15, sel = (lane >> 4) & 1;
    const __bf16* p = base + (long long)n * ld + sel * 8;
    Frag f;
    f.h[0] = *(const v8bf*)(p);
    f.h[1] = *(const v8bf*)(p + 16);
    return f.v;
}
__device__ __forceinline__ v8f wmma_bf16(v16bf a, v16bf b, v8f c) {
    return __builtin_amdgcn_wmma_f32_16x16x32_bf16(false, a, false, b, (short)0, c, false, false);
}

// ---------- device-wide split barrier (persistent kernel) ----------
__device__ __forceinline__ void grid_barrier(unsigned* cnt, unsigned target) {
    __builtin_amdgcn_fence(__ATOMIC_RELEASE, "agent");   // flush h/hs stores to device scope
    __syncthreads();
    if (threadIdx.x == 0) {
        __hip_atomic_fetch_add(cnt, 1u, __ATOMIC_RELEASE, __HIP_MEMORY_SCOPE_AGENT);
        unsigned v;
        do {
            __builtin_amdgcn_s_sleep(1);
            v = __hip_atomic_load(cnt, __ATOMIC_ACQUIRE, __HIP_MEMORY_SCOPE_AGENT);
        } while (v < target);
    }
    __syncthreads();
    __builtin_amdgcn_fence(__ATOMIC_ACQUIRE, "agent");   // invalidate L0 before reading new h
}

// ---------- one-time conversion kernels ----------
__global__ void k_cvt(const float* __restrict__ src, __bf16* __restrict__ dst, long long total) {
    long long i = (long long)blockIdx.x * blockDim.x + threadIdx.x;
    if (i < total) dst[i] = f2bf(src[i]);
}

// dst[n*K + k] = bf16(src[k*N + n])  (src is [K,N] row-major -> dst is [N,K])
__global__ void k_transpose_cvt(const float* __restrict__ src, __bf16* __restrict__ dst,
                                int K, int N) {
    long long i = (long long)blockIdx.x * blockDim.x + threadIdx.x;
    long long total = (long long)K * N;
    if (i >= total) return;
    int n = (int)(i / K), k = (int)(i % K);
    dst[i] = f2bf(src[(long long)k * N + n]);
}

// WrExtT [4096, 1056]: row n = [Wr[:,n] (1024) | Wi[:,n] (16) | zeros (16)]
__global__ void k_build_wrext(const float* __restrict__ Wr, const float* __restrict__ Wi,
                              __bf16* __restrict__ dst) {
    long long i = (long long)blockIdx.x * blockDim.x + threadIdx.x;
    const long long total = (long long)G4 * KEXT;
    if (i >= total) return;
    int n = (int)(i / KEXT), k = (int)(i % KEXT);
    float v;
    if (k < H)            v = Wr[(long long)k * G4 + n];
    else if (k < H + AD)  v = Wi[(long long)(k - H) * G4 + n];
    else                  v = 0.0f;
    dst[i] = f2bf(v);
}

// zero K-extension tails of both A buffers; stage action[:,0] into A0; zero grid barrier
__global__ void k_init_tail(const float* __restrict__ action,
                            __bf16* __restrict__ a0, __bf16* __restrict__ a1,
                            unsigned* __restrict__ gbar) {
    int i = blockIdx.x * blockDim.x + threadIdx.x;   // BS * 32
    if (i == 0) gbar[0] = 0u;
    if (i >= BS * 32) return;
    int b = i >> 5, j = i & 31;
    __bf16 z = f2bf(0.0f);
    a1[(long long)b * KEXT + H + j] = z;
    a0[(long long)b * KEXT + H + j] = (j < AD) ? f2bf(action[((long long)b * T + 0) * AD + j]) : z;
}

// ---------- init GEMMs: c0 = relu(flat@Wc+bc) ; h0 = relu(flat@Wh+bh) ----------
__global__ void __launch_bounds__(256) k_init_gemm(
        const __bf16* __restrict__ A,    // flat bf16 [256, 8192]
        const __bf16* __restrict__ WcT,  // [1024, 8192]
        const __bf16* __restrict__ WhT,  // [1024, 8192]
        const float*  __restrict__ bc,
        const float*  __restrict__ bh,
        float*  __restrict__ c_buf,      // [256, 1024] fp32
        __bf16* __restrict__ aext0)      // [256, 1056] bf16 (h0 -> cols 0..1023)
{
    const int tid  = threadIdx.x;
    const int wave = tid >> 5, lane = tid & 31;
    const int m0 = blockIdx.y * 32 + (wave & 1) * 16;
    const int n0 = blockIdx.x * 64 + (wave >> 1) * 16;
    const bool isH = (blockIdx.z == 1);
    const __bf16* BT = isH ? WhT : WcT;

    v8f acc = {};
    const __bf16* Abase = A  + (long long)m0 * HL_F;
    const __bf16* Bbase = BT + (long long)n0 * HL_F;
    for (int k = 0; k < HL_F; k += 32) {
        v16bf af = load_fragA(Abase + k, HL_F, lane);
        v16bf bf = load_fragB(Bbase + k, HL_F, lane);
        acc = wmma_bf16(af, bf, acc);
    }
    const int col  = n0 + (lane & 15);
    const float bias = (isH ? bh : bc)[col];
#pragma unroll
    for (int r = 0; r < 8; ++r) {
        int row = m0 + ((lane < 16) ? r : (r + 8));
        float v = acc[r] + bias;
        v = v > 0.0f ? v : 0.0f;
        if (isH) aext0[(long long)row * KEXT + col] = f2bf(v);
        else     c_buf[(long long)row * H    + col] = v;
    }
}

// ---------- persistent LSTM recurrence ----------
// grid = 128 blocks (rb = blk&3 -> 64 rows; cb = blk>>2 -> 32 hidden cols), 256 threads.
// B slice (128 gate-cols x 1056) staged in LDS once (TDM if available), reused 64 steps.
// Cell state c lives in registers for the whole recurrence; grid barrier between steps.
__global__ void __launch_bounds__(256) k_lstm_persist(
        __bf16* __restrict__ Aext0,        // [256, 1056] (h_t | action_t | 0) ping
        __bf16* __restrict__ Aext1,        // pong
        const __bf16* __restrict__ WrExtT, // [4096, 1056]
        const float*  __restrict__ bg,     // [4096]
        const float*  __restrict__ c_buf,  // [256, 1024] c0
        __bf16* __restrict__ hs,           // [256, 64, 1024]
        const float* __restrict__ action,  // [256, 64, 16]
        unsigned* __restrict__ gbar)
{
    extern __shared__ __bf16 lb[];         // [128][KEXT] bf16
    const int tid  = threadIdx.x;
    const int wave = tid >> 5, lane = tid & 31;
    const int rb = blockIdx.x & 3;         // rows   [rb*64,  +64)
    const int cb = blockIdx.x >> 2;        // h-cols [cb*32,  +32)

    // ---- stage B slice into LDS: gate g rows g*1024 + cb*32 .. +31 ----
#if defined(__gfx1250__) && __has_builtin(__builtin_amdgcn_tensor_load_to_lds)
    if (tid == 0) {
#pragma unroll
        for (int g = 0; g < 4; ++g) {
            unsigned long long ga =
                (unsigned long long)(WrExtT + ((long long)g * H + (long long)cb * 32) * KEXT);
            u32x4 g0 = {};
            g0[0] = 1u;                                         // count=1 user descriptor
            g0[1] = (unsigned)(g * 32 * KEXT * 2);              // lds_addr (bytes)
            g0[2] = (unsigned)(ga & 0xFFFFFFFFu);               // global_addr lo
            g0[3] = (unsigned)((ga >> 32) & 0x01FFFFFFu) | (2u << 30); // addr hi | type=2
            i32x8 g1 = {};
            g1[0] = 1 << 16;                                    // data_size = 2 bytes
            g1[1] = (int)((unsigned)(KEXT & 0xFFFF) << 16);     // tensor_dim0 lo16
            g1[2] = (int)((unsigned)(G4 & 0xFFFF) << 16);       // tensor_dim1 lo16
            g1[3] = (int)((unsigned)(KEXT & 0xFFFF) << 16);     // tile_dim0 = KEXT
            g1[4] = 32;                                         // tile_dim1 = 32 rows
            g1[5] = KEXT;                                       // tensor_dim0_stride
            i32x4 z4 = {};
            i32x8 z8 = {};
            __builtin_amdgcn_tensor_load_to_lds(g0, g1, z4, z4, z8, 0);
        }
    }
    __builtin_amdgcn_s_wait_tensorcnt((short)0);
#else
    {
        const int chunks = KEXT / 8;                            // 132 x 16B per row
        for (int idx = tid; idx < 128 * chunks; idx += 256) {
            int r = idx / chunks, c = idx % chunks;
            int grow = (r >> 5) * H + cb * 32 + (r & 31);
            *(v8bf*)(lb + (long long)r * KEXT + c * 8) =
                *(const v8bf*)(WrExtT + (long long)grow * KEXT + c * 8);
        }
    }
#endif
    __syncthreads();

    // ---- per-lane invariant coordinates ----
    const int m0  = rb * 64 + (wave & 3) * 16;     // row-tile base
    const int jh  = (wave >> 2) * 16;              // hidden-col half within block
    const int col = cb * 32 + jh + (lane & 15);    // global hidden col
    const float bi  = bg[0 * H + col];
    const float bf_ = bg[1 * H + col];
    const float bgg = bg[2 * H + col];
    const float bo_ = bg[3 * H + col];
    float creg[8];
#pragma unroll
    for (int r = 0; r < 8; ++r) {
        int row = m0 + ((lane < 16) ? r : (r + 8));
        creg[r] = c_buf[(long long)row * H + col];
    }
    const __bf16* lb0 = lb + (long long)(0 * 32 + jh) * KEXT;
    const __bf16* lb1 = lb + (long long)(1 * 32 + jh) * KEXT;
    const __bf16* lb2 = lb + (long long)(2 * 32 + jh) * KEXT;
    const __bf16* lb3 = lb + (long long)(3 * 32 + jh) * KEXT;
    __bf16* bufs[2] = { Aext0, Aext1 };

    for (int t = 0; t < T; ++t) {
        const __bf16* Acur = bufs[t & 1];
        __bf16*       Anext = bufs[(t + 1) & 1];
        const __bf16* Abase = Acur + (long long)m0 * KEXT;

        v8f a0 = {}, a1 = {}, a2 = {}, a3 = {};
        for (int k = 0; k < KEXT; k += 32) {
            v16bf af = load_fragA(Abase + k, KEXT, lane);
            v16bf b0 = load_fragB(lb0 + k, KEXT, lane);
            v16bf b1 = load_fragB(lb1 + k, KEXT, lane);
            v16bf b2 = load_fragB(lb2 + k, KEXT, lane);
            v16bf b3 = load_fragB(lb3 + k, KEXT, lane);
            a0 = wmma_bf16(af, b0, a0);
            a1 = wmma_bf16(af, b1, a1);
            a2 = wmma_bf16(af, b2, a2);
            a3 = wmma_bf16(af, b3, a3);
        }
#pragma unroll
        for (int r = 0; r < 8; ++r) {
            int row = m0 + ((lane < 16) ? r : (r + 8));
            float iv = sigmoid_f(a0[r] + bi);
            float fv = sigmoid_f(a1[r] + bf_);
            float gv = tanh_f  (a2[r] + bgg);
            float ov = sigmoid_f(a3[r] + bo_);
            float cn = fv * creg[r] + iv * gv;
            float hn = ov * tanh_f(cn);
            creg[r] = cn;
            __bf16 hb = f2bf(hn);
            Anext[(long long)row * KEXT + col] = hb;
            hs[((long long)row * T + t) * H + col] = hb;
        }
        if (t + 1 < T) {
            if (cb == 0) {   // blocks 0..3 stage action[:, t+1] for their 64 rows
                for (int e = tid; e < 64 * AD; e += 256) {
                    int rr = rb * 64 + (e >> 4);
                    int a  = e & 15;
                    Anext[(long long)rr * KEXT + H + a] =
                        f2bf(action[((long long)rr * T + (t + 1)) * AD + a]);
                }
            }
            grid_barrier(gbar, (unsigned)NBLK * (unsigned)(t + 1));
        }
    }
}

// ---------- final projection: out[b,t,o] = hs[b,t,:] @ Wo[:,o] + bo[o] ----------
__global__ void k_proj(const __bf16* __restrict__ hs, const float* __restrict__ Wo,
                       const float* __restrict__ bo, float* __restrict__ out) {
    int i = blockIdx.x * blockDim.x + threadIdx.x;
    if (i >= BS * T * OUTD) return;
    int o  = i & (OUTD - 1);
    int bt = i >> 4;
    const __bf16* h = hs + (long long)bt * H;
    float acc = bo[o];
#pragma unroll 4
    for (int k = 0; k < H; ++k) acc += bf2f(h[k]) * Wo[(long long)k * OUTD + o];
    out[i] = acc;
}

// ---------- host ----------
extern "C" void kernel_launch(void* const* d_in, const int* in_sizes, int n_in,
                              void* d_out, int out_size, void* d_ws, size_t ws_size,
                              hipStream_t stream) {
    (void)in_sizes; (void)n_in; (void)out_size; (void)ws_size;
    const float* history = (const float*)d_in[0];
    const float* action  = (const float*)d_in[1];
    const float* Wc = (const float*)d_in[2];
    const float* bc = (const float*)d_in[3];
    const float* Wh = (const float*)d_in[4];
    const float* bh = (const float*)d_in[5];
    const float* Wi = (const float*)d_in[6];
    const float* Wr = (const float*)d_in[7];
    const float* bg = (const float*)d_in[8];
    const float* Wo = (const float*)d_in[9];
    const float* bo = (const float*)d_in[10];
    float* out = (float*)d_out;

    size_t off = 0;
    auto alloc = [&](size_t bytes) -> void* {
        void* p = (char*)d_ws + off;
        off += (bytes + 255) & ~(size_t)255;
        return p;
    };
    __bf16*   flatA  = (__bf16*)alloc((size_t)BS * HL_F * 2);
    __bf16*   WcT    = (__bf16*)alloc((size_t)H * HL_F * 2);
    __bf16*   WhT    = (__bf16*)alloc((size_t)H * HL_F * 2);
    __bf16*   WrExtT = (__bf16*)alloc((size_t)G4 * KEXT * 2);
    __bf16*   Aext0  = (__bf16*)alloc((size_t)BS * KEXT * 2);
    __bf16*   Aext1  = (__bf16*)alloc((size_t)BS * KEXT * 2);
    float*    c_buf  = (float*) alloc((size_t)BS * H * 4);
    __bf16*   hs     = (__bf16*)alloc((size_t)BS * T * H * 2);
    unsigned* gbar   = (unsigned*)alloc(256);

    // one-time conversions / transposes
    {
        long long tot = (long long)BS * HL_F;
        k_cvt<<<dim3((unsigned)((tot + 255) / 256)), 256, 0, stream>>>(history, flatA, tot);
    }
    {
        long long tot = (long long)HL_F * H;
        k_transpose_cvt<<<dim3((unsigned)((tot + 255) / 256)), 256, 0, stream>>>(Wc, WcT, HL_F, H);
        k_transpose_cvt<<<dim3((unsigned)((tot + 255) / 256)), 256, 0, stream>>>(Wh, WhT, HL_F, H);
    }
    {
        long long tot = (long long)G4 * KEXT;
        k_build_wrext<<<dim3((unsigned)((tot + 255) / 256)), 256, 0, stream>>>(Wr, Wi, WrExtT);
    }
    k_init_tail<<<dim3((BS * 32 + 255) / 256), 256, 0, stream>>>(action, Aext0, Aext1, gbar);

    // c0 / h0
    k_init_gemm<<<dim3(16, 8, 2), 256, 0, stream>>>(flatA, WcT, WhT, bc, bh, c_buf, Aext0);

    // whole 64-step recurrence in ONE persistent kernel (grid barrier between steps)
    k_lstm_persist<<<dim3(NBLK), 256, LDSB_BYTES, stream>>>(
        Aext0, Aext1, WrExtT, bg, c_buf, hs, action, gbar);

    // output projection
    k_proj<<<dim3((BS * T * OUTD + 255) / 256), 256, 0, stream>>>(hs, Wo, bo, out);
}
